// TerrellPenroseSNN_45183055954004
// MI455X (gfx1250) — compile-verified, matrix operand order checked by reading, not run on hardware
//
#include <hip/hip_runtime.h>
#include <hip/hip_bf16.h>

typedef _Float16 h16;
typedef __attribute__((ext_vector_type(16))) _Float16 v16h;
typedef __attribute__((ext_vector_type(8)))  _Float16 v8h;
typedef __attribute__((ext_vector_type(8)))  float    v8f;

#define BETA 0.9f
#define THR1 1.0f
#define THR2 0.8f

enum { B_ = 256, T_ = 100, I_ = 1024, H_ = 2048, O_ = 1024 };

// ---------------------------------------------------------------------------
// Fragment loader for the CDNA5 16-bit 16x32 A/B layout (CDNA5 ISA
// §7.12.2): lane l (row = l%16, g = l/16) holds K = [8g .. 8g+7] in elems 0..7
// and K = [16+8g .. 16+8g+7] in elems 8..15 -> two aligned 16B loads.
// ---------------------------------------------------------------------------
__device__ __forceinline__ v16h ld_frag(const h16* __restrict__ p, int g) {
  union { struct { v8h lo, hi; } s; v16h v; } u;
  u.s.lo = *(const v8h*)(p + (g << 3));
  u.s.hi = *(const v8h*)(p + 16 + (g << 3));
  return u.v;
}

// ---------------------------------------------------------------------------
// One-time (per launch) setup: f32 -> f16 weights, delay factors, zero state.
// ---------------------------------------------------------------------------
__global__ void snn_init_kernel(const float* __restrict__ W1, const float* __restrict__ W2,
                                const float* __restrict__ cd1, const float* __restrict__ v1,
                                const float* __restrict__ cd2, const float* __restrict__ v2,
                                h16* __restrict__ W1h, h16* __restrict__ W2h,
                                float* __restrict__ df1, float* __restrict__ df2,
                                float* __restrict__ p1, float* __restrict__ p2,
                                float* __restrict__ accO) {
  const long tid = (long)blockIdx.x * blockDim.x + threadIdx.x;
  const long stride = (long)gridDim.x * blockDim.x;
  for (long i = tid; i < (long)H_ * I_; i += stride) W1h[i] = (h16)W1[i];
  for (long i = tid; i < (long)O_ * H_; i += stride) W2h[i] = (h16)W2[i];
  const float vc1 = fminf(fmaxf(v1[0], 0.0f), 0.999f);
  const float g1 = 1.0f / sqrtf(1.0f - vc1 * vc1);
  for (long i = tid; i < I_; i += stride) df1[i] = __expf(-(g1 * fabsf(cd1[i]) * vc1));
  const float vc2 = fminf(fmaxf(v2[0], 0.0f), 0.999f);
  const float g2 = 1.0f / sqrtf(1.0f - vc2 * vc2);
  for (long i = tid; i < H_; i += stride) df2[i] = __expf(-(g2 * fabsf(cd2[i]) * vc2));
  for (long i = tid; i < (long)B_ * I_; i += stride) p1[i] = 0.0f;
  for (long i = tid; i < (long)B_ * H_; i += stride) p2[i] = 0.0f;
  for (long i = tid; i < (long)B_ * O_; i += stride) accO[i] = 0.0f;
}

// ---------------------------------------------------------------------------
// Layer-1 LIF for t=0 only (later steps are fused into gemm2's epilogue).
// ---------------------------------------------------------------------------
__global__ void snn_lif1_kernel(const float* __restrict__ x, const float* __restrict__ df1,
                                float* __restrict__ p1, h16* __restrict__ s1h) {
  const int idx = blockIdx.x * blockDim.x + threadIdx.x;   // grid == B_*I_
  const int b = idx >> 10;            // I_ == 1024
  const int i = idx & (I_ - 1);
  const float xv = x[(long)b * T_ * I_ + i];               // t = 0
  const float p = p1[idx];                                 // == 0 from init
  const float np = p * BETA + xv * df1[i];
  const float s = (np > THR1) ? 1.0f : 0.0f;
  p1[idx] = np * (1.0f - s);
  s1h[idx] = (h16)s;
}

// ---------------------------------------------------------------------------
// GEMM1 (h = s1 @ W1^T + b1) with layer-2 LIF fused in the epilogue.
// 128 threads = 4 waves; wave tile = 16(M) x 64(N); K-loop step 32 (WMMA).
// ---------------------------------------------------------------------------
__global__ __launch_bounds__(128, 2) void snn_gemm1_lif2_kernel(
    const h16* __restrict__ A,      // s1h [B_, I_]
    const h16* __restrict__ W,      // W1h [H_, I_]
    const float* __restrict__ bias, // b1 [H_]
    const float* __restrict__ df,   // df2 [H_]
    float* __restrict__ pot,        // p2  [B_, H_]
    h16* __restrict__ S) {          // s2h [B_, H_]
  const int K = I_, N = H_;
  const int lane = threadIdx.x & 31;
  const int wave = threadIdx.x >> 5;
  const int nl = lane & 15;
  const int g = lane >> 4;
  const int m_base = blockIdx.y << 4;
  const int n_base = ((blockIdx.x << 2) + wave) << 6;

  const h16* __restrict__ ap = A + (long)(m_base + nl) * K;
  const h16* __restrict__ wp = W + (long)(n_base + nl) * K;

  v8f acc[4] = {};
#pragma unroll 2
  for (int k0 = 0; k0 < K; k0 += 32) {
    const v16h a = ld_frag(ap + k0, g);
#pragma unroll
    for (int j = 0; j < 4; ++j) {
      const v16h b = ld_frag(wp + (long)(j << 4) * K + k0, g);
      acc[j] = __builtin_amdgcn_wmma_f32_16x16x32_f16(
          false, a, false, b, (short)0, acc[j], false, false);
    }
  }
  // Epilogue: h = acc + b1[col]; fused LIF2 (each (m,col) touched exactly once).
#pragma unroll
  for (int j = 0; j < 4; ++j) {
    const int col = n_base + (j << 4) + nl;
    const float bj = bias[col];
    const float dj = df[col];
#pragma unroll
    for (int e = 0; e < 8; ++e) {
      const int m = m_base + (g << 3) + e;     // C/D layout: M = e + 8*g
      const long off = (long)m * N + col;
      const float h = acc[j][e] + bj;
      const float po = pot[off];
      const float np = po * BETA + h * dj;
      const float s = (np > THR2) ? 1.0f : 0.0f;
      pot[off] = np * (1.0f - s);
      S[off] = (h16)s;
    }
  }
}

// ---------------------------------------------------------------------------
// GEMM2 (o = s2 @ W2^T + b2); epilogue:
//   1) accO += tw[t] * o  (tw from closed-form geometric series of v1)
//   2) fused LIF1 for step t+1 (block's [16 x 256] O-tile maps 1:1 onto the
//      same [16 x 256] region of [B_, I_] since O_ == I_; each p1/s1 element
//      is owned by exactly one block -> race-free, no ping-pong needed)
//   3) on the final step, emit out = accO * out_scale + out_bias directly
// ---------------------------------------------------------------------------
__global__ __launch_bounds__(128, 2) void snn_gemm2_fused_kernel(
    const h16* __restrict__ A,      // s2h [B_, H_]
    const h16* __restrict__ W,      // W2h [O_, H_]
    const float* __restrict__ bias, // b2 [O_]
    const float* __restrict__ v1,
    float* __restrict__ accO,       // [B_, O_]
    const float* __restrict__ x,    // [B_, T_, I_]
    const float* __restrict__ df1,  // [I_]
    float* __restrict__ p1,         // [B_, I_]
    h16* __restrict__ s1h,          // [B_, I_]
    const float* __restrict__ sc,   // out_scale [O_]
    const float* __restrict__ bs,   // out_bias  [O_]
    float* __restrict__ out,        // [B_, O_]
    int t) {
  const int K = H_, N = O_;
  const int lane = threadIdx.x & 31;
  const int wave = threadIdx.x >> 5;
  const int nl = lane & 15;
  const int g = lane >> 4;
  const int m_base = blockIdx.y << 4;
  const int n_base = ((blockIdx.x << 2) + wave) << 6;

  const h16* __restrict__ ap = A + (long)(m_base + nl) * K;
  const h16* __restrict__ wp = W + (long)(n_base + nl) * K;

  v8f acc[4] = {};
#pragma unroll 2
  for (int k0 = 0; k0 < K; k0 += 32) {
    const v16h a = ld_frag(ap + k0, g);
#pragma unroll
    for (int j = 0; j < 4; ++j) {
      const v16h b = ld_frag(wp + (long)(j << 4) * K + k0, g);
      acc[j] = __builtin_amdgcn_wmma_f32_16x16x32_f16(
          false, a, false, b, (short)0, acc[j], false, false);
    }
  }

  // Temporal weight tw[t] = r^t / sum_{u<T} r^u, r = exp(-(gamma-1)).
  const float vc = fminf(fmaxf(v1[0], 0.0f), 0.999f);
  const float gam = 1.0f / sqrtf(1.0f - vc * vc);
  const float lam = gam - 1.0f;
  const float r = __expf(-lam);
  const float denom = (fabsf(lam) > 1e-12f)
                          ? (1.0f - __powf(r, (float)T_)) / (1.0f - r)
                          : (float)T_;
  const float tw = __expf(-lam * (float)t) / denom;

  const bool last = (t == T_ - 1);
#pragma unroll
  for (int j = 0; j < 4; ++j) {
    const int col = n_base + (j << 4) + nl;
    const float bj = bias[col];
#pragma unroll
    for (int e = 0; e < 8; ++e) {
      const int m = m_base + (g << 3) + e;
      const long off = (long)m * N + col;
      const float v = accO[off] + tw * (acc[j][e] + bj);
      accO[off] = v;
      if (last) out[off] = v * sc[col] + bs[col];   // uniform branch
    }
  }

  // Fused LIF1 for step t+1 over this block's private [16 x 256] tile of
  // [B_, I_] (linear thread mapping, coalesced over columns).
  if (t + 1 < T_) {                                 // uniform branch
    const int i_base = blockIdx.x << 8;             // 256 cols per block
    const int tid = threadIdx.x;
#pragma unroll
    for (int q = 0; q < 32; ++q) {
      const int local = (q << 7) + tid;             // 0 .. 4095
      const int row = local >> 8;
      const int colI = local & 255;
      const int m = m_base + row;
      const int i = i_base + colI;
      const long pidx = (long)m * I_ + i;
      const float xv = x[(long)m * T_ * I_ + (long)(t + 1) * I_ + i];
      const float p = p1[pidx];
      const float np = p * BETA + xv * df1[i];
      const float s = (np > THR1) ? 1.0f : 0.0f;
      p1[pidx] = np * (1.0f - s);
      s1h[pidx] = (h16)s;
    }
  }
}

// ---------------------------------------------------------------------------
extern "C" void kernel_launch(void* const* d_in, const int* in_sizes, int n_in,
                              void* d_out, int out_size, void* d_ws, size_t ws_size,
                              hipStream_t stream) {
  (void)in_sizes; (void)n_in; (void)out_size; (void)ws_size;
  const float* x   = (const float*)d_in[0];
  const float* cd1 = (const float*)d_in[1];
  const float* v1  = (const float*)d_in[2];
  const float* cd2 = (const float*)d_in[3];
  const float* v2  = (const float*)d_in[4];
  const float* W1  = (const float*)d_in[5];
  const float* b1  = (const float*)d_in[6];
  const float* W2  = (const float*)d_in[7];
  const float* b2  = (const float*)d_in[8];
  const float* osc = (const float*)d_in[9];
  const float* obs = (const float*)d_in[10];
  float* out = (float*)d_out;

  char* ws = (char*)d_ws;
  size_t off = 0;
  auto take = [&](size_t bytes) -> char* {
    char* p = ws + off;
    off = (off + bytes + 255) & ~(size_t)255;
    return p;
  };
  h16*   W1h  = (h16*)  take((size_t)H_ * I_ * sizeof(h16));   // 4 MB
  h16*   W2h  = (h16*)  take((size_t)O_ * H_ * sizeof(h16));   // 4 MB
  h16*   s1h  = (h16*)  take((size_t)B_ * I_ * sizeof(h16));   // 512 KB
  h16*   s2h  = (h16*)  take((size_t)B_ * H_ * sizeof(h16));   // 1 MB
  float* p1   = (float*)take((size_t)B_ * I_ * sizeof(float)); // 1 MB
  float* p2   = (float*)take((size_t)B_ * H_ * sizeof(float)); // 2 MB
  float* accO = (float*)take((size_t)B_ * O_ * sizeof(float)); // 1 MB
  float* df1  = (float*)take((size_t)I_ * sizeof(float));
  float* df2  = (float*)take((size_t)H_ * sizeof(float));

  snn_init_kernel<<<2048, 256, 0, stream>>>(W1, W2, cd1, v1, cd2, v2,
                                            W1h, W2h, df1, df2, p1, p2, accO);
  snn_lif1_kernel<<<(B_ * I_) / 256, 256, 0, stream>>>(x, df1, p1, s1h);

  for (int t = 0; t < T_; ++t) {
    snn_gemm1_lif2_kernel<<<dim3(H_ / 256, B_ / 16), 128, 0, stream>>>(
        s1h, W1h, b1, df2, p2, s2h);
    snn_gemm2_fused_kernel<<<dim3(O_ / 256, B_ / 16), 128, 0, stream>>>(
        s2h, W2h, b2, v1, accO, x, df1, p1, s1h, osc, obs, out, t);
  }
}
